// BERT_CRF_NER_38405597560878
// MI455X (gfx1250) — matrix operand order repaired and checked by reading, not compiled, not run on hardware
//
#include <hip/hip_runtime.h>
#include <hip/hip_bf16.h>

typedef __attribute__((ext_vector_type(16))) _Float16 v16h;
typedef __attribute__((ext_vector_type(8)))  float    v8f;

#define Bdim 128
#define Sdim 256
#define Hdim 768
#define Ldim 24
#define LPAD 32
#define KSTEPS (Hdim / 32)   // 24
#define WAVES 8

// ---------------------------------------------------------------------------
// Kernel 1: per-batch valid-token count (masks are contiguous prefixes).
// ---------------------------------------------------------------------------
__global__ void ner_lengths_kernel(const int* __restrict__ mask,
                                   int* __restrict__ nvalid) {
    int b = threadIdx.x;
    if (b < Bdim) {
        const int* row = mask + b * Sdim;
        int s = 0;
        for (int i = 0; i < Sdim; ++i) s += row[i];
        nvalid[b] = s;
    }
}

// ---------------------------------------------------------------------------
// Kernel 2: fused compact (shift-by-one, truncate at len-2) + GEMM (WMMA f16)
//           + bias + sigmoid. One 16-row M-tile per wave; K=768 in 24 steps;
//           L=24 split into two 16-wide N-tiles (cols 24..31 zeroed).
// W is staged once per block into LDS in B-fragment order (f16).
// ---------------------------------------------------------------------------
__global__ void __launch_bounds__(256) ner_emissions_kernel(
    const float* __restrict__ tf,      // [B,S,H]
    const int*   __restrict__ nvalid,  // [B]
    const float* __restrict__ W,       // [H,L]
    const float* __restrict__ bias,    // [L]
    float*       __restrict__ em) {    // [B*S, LPAD]
    __shared__ v16h sWB[KSTEPS * 2 * 32];   // 48 KB of f16 B-fragments
    _Float16* sWBh = (_Float16*)sWB;

    int tid = threadIdx.x;
    // B-fragment layout for 16-bit B (32x16):
    //   lane l: column N = l&15, K-half = l>>4, element e -> K = ks*32 + (e<8 ? half*8 + e : 16 + half*8 + e-8)
    for (int idx = tid; idx < KSTEPS * 2 * 32 * 16; idx += 256) {
        int e    = idx & 15;
        int lane = (idx >> 4) & 31;
        int tile = (idx >> 9) & 1;
        int ks   = idx >> 10;
        int k    = ks * 32 + (lane >> 4) * 8 + (e & 7) + (e >> 3) * 16;
        int col  = tile * 16 + (lane & 15);
        float v  = (col < Ldim) ? W[k * Ldim + col] : 0.0f;
        sWBh[idx] = (_Float16)v;
    }
    __syncthreads();

    int lane  = tid & 31;
    int wid   = tid >> 5;
    int mtile = blockIdx.x * WAVES + wid;       // 0..2047

    // A-fragment (16-bit A 16x32): lanes 0-15: M=lane, K chunks {0..7},{16..23};
    //                              lanes16-31: M=lane-16, K chunks {8..15},{24..31}
    int m     = lane & 15;
    int khalf = lane >> 4;
    int g     = mtile * 16 + m;
    int bb    = g / Sdim;
    int j     = g - bb * Sdim;
    int nv    = nvalid[bb];
    bool valid = (j < nv - 2);
    int  jj    = valid ? (j + 1) : 0;           // compact == shift-left-by-one
    float vs   = valid ? 1.0f : 0.0f;
    const float* src = tf + ((size_t)bb * Sdim + jj) * Hdim;

    v8f c0 = {};
    v8f c1 = {};
    for (int ks = 0; ks < KSTEPS; ++ks) {
        int k0 = ks * 32;
        const float4* p = (const float4*)(src + k0 + khalf * 8);
        const float4* q = (const float4*)(src + k0 + 16 + khalf * 8);
        float4 f0 = p[0], f1 = p[1], f2 = q[0], f3 = q[1];
        v16h a;
        a[0]=(_Float16)(f0.x*vs); a[1]=(_Float16)(f0.y*vs); a[2]=(_Float16)(f0.z*vs); a[3]=(_Float16)(f0.w*vs);
        a[4]=(_Float16)(f1.x*vs); a[5]=(_Float16)(f1.y*vs); a[6]=(_Float16)(f1.z*vs); a[7]=(_Float16)(f1.w*vs);
        a[8]=(_Float16)(f2.x*vs); a[9]=(_Float16)(f2.y*vs); a[10]=(_Float16)(f2.z*vs); a[11]=(_Float16)(f2.w*vs);
        a[12]=(_Float16)(f3.x*vs); a[13]=(_Float16)(f3.y*vs); a[14]=(_Float16)(f3.z*vs); a[15]=(_Float16)(f3.w*vs);

        v16h b0 = sWB[(ks * 2 + 0) * 32 + lane];
        v16h b1 = sWB[(ks * 2 + 1) * 32 + lane];
        c0 = __builtin_amdgcn_wmma_f32_16x16x32_f16(false, a, false, b0, (short)0, c0, false, false);
        c1 = __builtin_amdgcn_wmma_f32_16x16x32_f16(false, a, false, b1, (short)0, c1, false, false);
    }

    // C/D layout: lanes 0-15 hold M=v, lanes 16-31 hold M=8+v; N=lane&15.
    int N  = lane & 15;
    int mo = (lane >> 4) * 8;
    float bias0 = bias[N];
    float bias1 = (16 + N < Ldim) ? bias[16 + N] : 0.0f;
    for (int v = 0; v < 8; ++v) {
        int gr = mtile * 16 + mo + v;
        float x0 = c0[v] + bias0;
        em[(size_t)gr * LPAD + N] = 1.0f / (1.0f + __expf(-x0));
        if (16 + N < Ldim) {
            float x1 = c1[v] + bias1;
            em[(size_t)gr * LPAD + 16 + N] = 1.0f / (1.0f + __expf(-x1));
        }
    }
}

// ---------------------------------------------------------------------------
// Kernel 3: masked Viterbi + backtrace. One wave per batch, lane = tag.
//  - transitions column held in registers (24 VGPRs / lane)
//  - score broadcast via v_readlane (__shfl with constant lane)
//  - depth-5 tournament argmax with first-max tie-break (left wins ties)
//  - emissions staged once into LDS (coalesced float4 copy)
//  - contiguous-mask shortcut: run recurrence only for step < nlab
// ---------------------------------------------------------------------------
__global__ void __launch_bounds__(32) ner_viterbi_kernel(
    const float* __restrict__ em,       // [B*S, LPAD]
    const int*   __restrict__ tlmask,   // [B,S]
    const float* __restrict__ trans,    // [L,L]
    const float* __restrict__ start_t,  // [L]
    const float* __restrict__ end_t,    // [L]
    int*         __restrict__ path) {   // [B,S]
    __shared__ float sEm[Sdim * LPAD];              // 32 KB
    __shared__ unsigned char sBps[(Sdim - 1) * Ldim];
    __shared__ float sFinal[Ldim];
    __shared__ int   sNlab;

    int b  = blockIdx.x;
    int t  = threadIdx.x;
    int tc = (t < Ldim) ? t : (Ldim - 1);           // clamp for lanes 24..31

    // Stage this batch's emissions into LDS (fully coalesced).
    const float4* src4 = (const float4*)(em + (size_t)b * Sdim * LPAD);
    float4*       dst4 = (float4*)sEm;
    for (int i = t; i < (Sdim * LPAD) / 4; i += 32) dst4[i] = src4[i];

    // nlab = sum(true_label_mask row) -- contiguous prefix length.
    int part = 0;
    for (int i = t; i < Sdim; i += 32) part += tlmask[b * Sdim + i];
    #pragma unroll
    for (int off = 16; off > 0; off >>= 1) part += __shfl_down(part, off, 32);
    if (t == 0) sNlab = part;

    // Transition column for this tag in registers.
    float tr[Ldim];
    #pragma unroll
    for (int p = 0; p < Ldim; ++p) tr[p] = trans[p * Ldim + tc];

    __syncthreads();
    int nlab = sNlab;

    float score = start_t[tc] + sEm[tc];            // time 0

    for (int step = 1; step < nlab; ++step) {
        float emt = sEm[step * LPAD + tc];
        // candidates: score[p] (broadcast) + trans[p][t]
        float val[Ldim];
        int   idx[Ldim];
        #pragma unroll
        for (int p = 0; p < Ldim; ++p) {
            val[p] = __shfl(score, p, 32) + tr[p];
            idx[p] = p;
        }
        // contiguous tournament; strict '>' picks right => ties keep smaller index
        #pragma unroll
        for (int n = Ldim; n > 1; n = (n + 1) >> 1) {
            int h = n >> 1;
            #pragma unroll
            for (int i = 0; i < h; ++i) {
                bool gt = val[2 * i + 1] > val[2 * i];
                val[i] = gt ? val[2 * i + 1] : val[2 * i];
                idx[i] = gt ? idx[2 * i + 1] : idx[2 * i];
            }
            if (n & 1) { val[h] = val[n - 1]; idx[h] = idx[n - 1]; }
        }
        score = val[0] + emt;
        if (t < Ldim) sBps[(step - 1) * Ldim + t] = (unsigned char)idx[0];
    }
    // padded tail: identity backpointers, score frozen
    if (t < Ldim) {
        int s0 = (nlab > 1) ? nlab : 1;
        for (int step = s0; step < Sdim; ++step)
            sBps[(step - 1) * Ldim + t] = (unsigned char)t;
        sFinal[t] = score + end_t[tc];
    }
    __syncthreads();

    if (t == 0) {
        float best = -3.0e38f;
        int tag = 0;
        for (int p = 0; p < Ldim; ++p)
            if (sFinal[p] > best) { best = sFinal[p]; tag = p; }   // first argmax
        for (int step = Sdim - 1; step >= 1; --step) {
            path[b * Sdim + step] = tag;
            tag = sBps[(step - 1) * Ldim + tag];
        }
        path[b * Sdim + 0] = tag;
    }
}

// ---------------------------------------------------------------------------
extern "C" void kernel_launch(void* const* d_in, const int* in_sizes, int n_in,
                              void* d_out, int out_size, void* d_ws, size_t ws_size,
                              hipStream_t stream) {
    (void)in_sizes; (void)n_in; (void)out_size; (void)ws_size;
    const float* tf      = (const float*)d_in[0];  // token_features [B,S,H]
    const int*   imask   = (const int*)  d_in[1];  // input_mask [B,S]
    const int*   tlmask  = (const int*)  d_in[2];  // true_label_mask [B,S]
    const float* W       = (const float*)d_in[3];  // [H,L]
    const float* bias    = (const float*)d_in[4];  // [L]
    const float* trans   = (const float*)d_in[5];  // [L,L]
    const float* start_t = (const float*)d_in[6];  // [L]
    const float* end_t   = (const float*)d_in[7];  // [L]
    int* path = (int*)d_out;                       // [B,S] int32 tags

    char*  ws     = (char*)d_ws;
    int*   nvalid = (int*)ws;                       // 128 ints
    float* em     = (float*)(ws + 1024);            // [32768, LPAD] floats (~4 MB)

    ner_lengths_kernel<<<1, 128, 0, stream>>>(imask, nvalid);

    const int mtiles = (Bdim * Sdim) / 16;          // 2048
    ner_emissions_kernel<<<mtiles / WAVES, 256, 0, stream>>>(tf, nvalid, W, bias, em);

    ner_viterbi_kernel<<<Bdim, 32, 0, stream>>>(em, tlmask, trans, start_t, end_t, path);
}
